// VariationalQuantumCircuit_41764261986826
// MI455X (gfx1250) — matrix operand order repaired and checked by reading, not compiled, not run on hardware
//
#include <hip/hip_runtime.h>
#include <hip/hip_bf16.h>
#include <math.h>

typedef __attribute__((ext_vector_type(16))) _Float16 v16h;
typedef __attribute__((ext_vector_type(8)))  _Float16 v8h;
typedef __attribute__((ext_vector_type(8)))  float    v8f;

#define D 128
#define QN 8
#define NPARAM 50
#define ROWS 128              // rows per block
#define LN_EPS 1e-5f
#define SF_STRIDE 132         // floats (16B aligned rows, conflict-reducing)
#define SW_STRIDE 136         // halves (16B aligned rows)

// ---------------- prep kernel: K scalar, u/v vectors, W2^T in f16 ----------------
__global__ __launch_bounds__(256) void vqc_prep(
    const float* __restrict__ vparams,   // [50]
    const float* __restrict__ W1,        // [8,128] row-major
    const float* __restrict__ W2,        // [128,128] row-major [k][n]
    _Float16* __restrict__ ws_w2t,       // [128,128]: w2t[n][k] = W2[k][n]
    float* __restrict__ ws_u,            // [128]
    float* __restrict__ ws_v,            // [128]
    float* __restrict__ ws_K)            // [1]
{
    int tid = threadIdx.x;
    // transpose + convert W2 -> f16 (tiny, once)
    for (int i = tid; i < D * D; i += 256) {
        int n = i >> 7, k = i & 127;
        ws_w2t[n * D + k] = (_Float16)W2[k * D + n];
    }
    if (tid < D) {
        float u = 0.f;
        #pragma unroll
        for (int q = 0; q < QN - 1; ++q) u += W1[q * D + tid];
        ws_u[tid] = u;
        ws_v[tid] = W1[(QN - 1) * D + tid];
    }
    if (tid == 0) {
        // op i is 'rot' iff i % 15 < 8 (DEPTH=6,Q=8 schedule, truncated at 50)
        float K = 1.f;
        for (int i = 0; i < NPARAM; ++i) {
            float p = vparams[i];
            if ((i % 15) < 8) { float c = cosf(p); K *= c * c; }
            else              { float f = 1.f - 0.1f * tanhf(p); K *= f * f; }
        }
        ws_K[0] = K;
    }
}

// ---------------- fused main kernel ----------------
__global__ __launch_bounds__(256) void vqc_main(
    const float* __restrict__ features,  // [B,128]
    const float* __restrict__ W_enc,     // [128,8]
    const float* __restrict__ b_enc,     // [8]
    const float* __restrict__ b1,        // [128]
    const float* __restrict__ b2,        // [128]
    const float* __restrict__ ln_w,      // [128]
    const float* __restrict__ ln_b,      // [128]
    const _Float16* __restrict__ ws_w2t, // [128,128] transposed f16 W2
    const float* __restrict__ ws_u,
    const float* __restrict__ ws_v,
    const float* __restrict__ ws_K,
    float* __restrict__ out)             // [B,128]
{
    __shared__ float    sF[ROWS * SF_STRIDE];       // features tile (f32)
    __shared__ _Float16 sW[D * SW_STRIDE];          // W2^T tile (f16)
    __shared__ float    sEvL[ROWS], sEvH[ROWS];
    __shared__ float    swenc[D], su[D], sv[D], sb1[D], sb2[D], slw[D], slb[D];

    const int tid  = threadIdx.x;
    const int r0   = blockIdx.x * ROWS;

    // ---- cooperative loads ----
    {   // features tile, fully coalesced float4
        const float4* gF = (const float4*)(features + (size_t)r0 * D);
        for (int i = tid; i < ROWS * (D / 4); i += 256) {
            int row = i >> 5, c4 = i & 31;
            float4 f = gF[row * (D / 4) + c4];
            *(float4*)&sF[row * SF_STRIDE + c4 * 4] = f;
        }
        // W2^T tile, 16B chunks
        for (int i = tid; i < D * (D / 8); i += 256) {
            int n = i >> 4, c8 = i & 15;
            v8h t = *(const v8h*)(ws_w2t + n * D + c8 * 8);
            *(v8h*)&sW[n * SW_STRIDE + c8 * 8] = t;
        }
        if (tid < D) {
            swenc[tid] = W_enc[tid * QN];   // column 0 of W_enc
            su[tid] = ws_u[tid];  sv[tid] = ws_v[tid];
            sb1[tid] = b1[tid];   sb2[tid] = b2[tid];
            slw[tid] = ln_w[tid]; slb[tid] = ln_b[tid];
        }
    }
    __syncthreads();

    // ---- per-row encode + collapsed circuit -> (ev_low, ev_hi) ----
    if (tid < ROWS) {
        float acc = 0.f;
        const float4* frow = (const float4*)&sF[tid * SF_STRIDE];
        const float4* wrow = (const float4*)swenc;
        #pragma unroll
        for (int i = 0; i < D / 4; ++i) {
            float4 f = frow[i], w = wrow[i];
            acc = fmaf(f.x, w.x, fmaf(f.y, w.y, fmaf(f.z, w.z, fmaf(f.w, w.w, acc))));
        }
        float K   = ws_K[0];
        float enc = tanhf(acc + b_enc[0]) * 3.14159265358979323846f;
        float h   = 0.5f * enc;
        float c   = cosf(h), s = sinf(h);
        float p0  = K * c * c;
        float p1  = s * s;
        sEvL[tid] = 2.f * p0 - 1.f;
        sEvH[tid] = 2.f * (p0 + p1) - 1.f;
    }
    __syncthreads();

    // ---- GEMM: relu(ev_l*u + ev_h*v + b1) @ W2, via WMMA f16->f32 ----
    const int wave = tid >> 5;           // 0..7 -> M tile
    const int lane = tid & 31;
    const int col0 = lane & 15;
    const int hiH  = lane >> 4;          // half-wave selector

    // A fragments (16x32 f16), built in registers from the rank-2 form.
    // Lane<16: rows 0-15, K = kb+{0..7, 16..23}; lane>=16: K = kb+{8..15, 24..31}.
    const int rA   = wave * 16 + col0;   // this lane's A row within tile
    const float evl = sEvL[rA], evh = sEvH[rA];
    const int  kSh  = hiH * 8;

    v16h afrag[4];
    #pragma unroll
    for (int kt = 0; kt < 4; ++kt) {
        int kb = kt * 32;
        v16h a;
        #pragma unroll
        for (int j = 0; j < 8; ++j) {
            int k1 = kb + kSh + j;
            int k2 = kb + 16 + kSh + j;
            float x1 = fmaf(evl, su[k1], fmaf(evh, sv[k1], sb1[k1]));
            float x2 = fmaf(evl, su[k2], fmaf(evh, sv[k2], sb1[k2]));
            a[j]     = (_Float16)fmaxf(x1, 0.f);
            a[j + 8] = (_Float16)fmaxf(x2, 0.f);
        }
        afrag[kt] = a;
    }

    // 8 N-tiles x 4 K-steps of v_wmma_f32_16x16x32_f16
    v8f accs[8];
    #pragma unroll
    for (int nt = 0; nt < 8; ++nt) {
        v8f acc = {};
        const int n = nt * 16 + col0;
        #pragma unroll
        for (int kt = 0; kt < 4; ++kt) {
            // B fragment (32x16): lane holds column n, K = kt*32 + hiH*8 + {0..7, 16..23}
            const _Float16* bp = &sW[n * SW_STRIDE + kt * 32 + hiH * 8];
            v8h lo = *(const v8h*)(bp);
            v8h hi = *(const v8h*)(bp + 16);
            v16h b;
            #pragma unroll
            for (int j = 0; j < 8; ++j) { b[j] = lo[j]; b[j + 8] = hi[j]; }
            acc = __builtin_amdgcn_wmma_f32_16x16x32_f16(
                false, afrag[kt], false, b, (short)0, acc, false, false);
        }
        accs[nt] = acc;
    }

    // ---- +b2, LayerNorm (half-wave reductions), residual, store ----
    // C/D layout: VGPR r, lanes 0-15 -> row r, col=lane; lanes 16-31 -> row r+8.
    float b2v[8];
    #pragma unroll
    for (int nt = 0; nt < 8; ++nt) b2v[nt] = sb2[nt * 16 + col0];

    const int mBase = wave * 16 + hiH * 8;
    #pragma unroll
    for (int r = 0; r < 8; ++r) {
        float s = 0.f, q = 0.f;
        #pragma unroll
        for (int nt = 0; nt < 8; ++nt) {
            float x = accs[nt][r] + b2v[nt];
            s += x; q += x * x;
        }
        #pragma unroll
        for (int off = 1; off < 16; off <<= 1) {
            s += __shfl_xor(s, off, 32);
            q += __shfl_xor(q, off, 32);
        }
        float mean = s * (1.f / 128.f);
        float var  = q * (1.f / 128.f) - mean * mean;
        float rstd = rsqrtf(var + LN_EPS);
        const int m    = mBase + r;
        const size_t grow = (size_t)(r0 + m) * D;
        #pragma unroll
        for (int nt = 0; nt < 8; ++nt) {
            int n   = nt * 16 + col0;
            float x = accs[nt][r] + b2v[nt];
            float y = (x - mean) * rstd * slw[n] + slb[n] + sF[m * SF_STRIDE + n];
            out[grow + n] = y;
        }
    }
}

extern "C" void kernel_launch(void* const* d_in, const int* in_sizes, int n_in,
                              void* d_out, int out_size, void* d_ws, size_t ws_size,
                              hipStream_t stream) {
    const float* features = (const float*)d_in[0];
    const float* W_enc    = (const float*)d_in[1];
    const float* b_enc    = (const float*)d_in[2];
    const float* vparams  = (const float*)d_in[3];
    const float* W1       = (const float*)d_in[4];
    const float* b1       = (const float*)d_in[5];
    const float* W2       = (const float*)d_in[6];
    const float* b2       = (const float*)d_in[7];
    const float* ln_w     = (const float*)d_in[8];
    const float* ln_b     = (const float*)d_in[9];
    float* out = (float*)d_out;

    _Float16* ws_w2t = (_Float16*)d_ws;                        // 32768 B
    float*    ws_u   = (float*)((char*)d_ws + 32768);          // 512 B
    float*    ws_v   = ws_u + D;                               // 512 B
    float*    ws_K   = ws_v + D;                               // 4 B

    hipLaunchKernelGGL(vqc_prep, dim3(1), dim3(256), 0, stream,
                       vparams, W1, W2, ws_w2t, ws_u, ws_v, ws_K);

    const int B = in_sizes[0] / D;      // 262144
    hipLaunchKernelGGL(vqc_main, dim3(B / ROWS), dim3(256), 0, stream,
                       features, W_enc, b_enc, b1, b2, ln_w, ln_b,
                       ws_w2t, ws_u, ws_v, ws_K, out);
}
